// transformer_layer_53652731462153
// MI455X (gfx1250) — matrix operand order repaired and checked by reading, compiled, not run on hardware
//
#include <hip/hip_runtime.h>
#include <cstdint>

typedef __bf16 bf16_t;
typedef bf16_t v8bf  __attribute__((ext_vector_type(8)));
typedef bf16_t v16bf __attribute__((ext_vector_type(16)));
typedef float  v8f   __attribute__((ext_vector_type(8)));

#define S_LEN  1194
#define BATCH  8
#define NHEAD  2
#define DMODEL 512
#define HDIM   256
#define FFDIM  2048
#define MROWS  (BATCH * S_LEN)   // 9552
#define MPAD   9600              // 75 * 128
#define SQP    1280              // padded sequence (10 * 128)
#define BHN    (BATCH * NHEAD)   // 16

// ---------------------------------------------------------------------------
// CDNA5 async global->LDS copy (ASYNCcnt-tracked, bypasses VGPRs)
// ---------------------------------------------------------------------------
__device__ inline void async_ld_b128(void* lds_ptr, const void* gptr) {
  // low 32 bits of a generic LDS pointer == workgroup-relative LDS address
  uint32_t l = (uint32_t)(uintptr_t)lds_ptr;
  asm volatile("global_load_async_to_lds_b128 %0, %1, off"
               :: "v"(l), "v"(gptr)
               : "memory");
}

#if defined(__has_builtin) && __has_builtin(__builtin_amdgcn_s_wait_asynccnt)
#define ASYNC_WAIT(n) __builtin_amdgcn_s_wait_asynccnt(n)
#else
#define ASYNC_WAIT(n) asm volatile("s_wait_asynccnt %0" :: "i"(n) : "memory")
#endif

// ---------------------------------------------------------------------------
// helpers
// ---------------------------------------------------------------------------
__device__ inline v8f wmma_bf16(v16bf a, v16bf b, v8f c) {
  // D = A(16x32 bf16) x B(32x16 bf16) + C(16x16 f32)
  return __builtin_amdgcn_wmma_f32_16x16x32_bf16(false, a, false, b, (short)0, c,
                                                 false, false);
}

__device__ inline v16bf ldpair(const bf16_t* p0, const bf16_t* p1) {
  v8bf lo = *(const v8bf*)p0;
  v8bf hi = *(const v8bf*)p1;
  return __builtin_shufflevector(lo, hi, 0, 1, 2, 3, 4, 5, 6, 7,
                                 8, 9, 10, 11, 12, 13, 14, 15);
}

// reductions within each 16-lane half of a wave32 (xor masks 1..8 stay in-half)
__device__ inline float half16_max(float v) {
  v = fmaxf(v, __shfl_xor(v, 1, 32));
  v = fmaxf(v, __shfl_xor(v, 2, 32));
  v = fmaxf(v, __shfl_xor(v, 4, 32));
  v = fmaxf(v, __shfl_xor(v, 8, 32));
  return v;
}
__device__ inline float half16_sum(float v) {
  v += __shfl_xor(v, 1, 32);
  v += __shfl_xor(v, 2, 32);
  v += __shfl_xor(v, 4, 32);
  v += __shfl_xor(v, 8, 32);
  return v;
}

// ---------------------------------------------------------------------------
// generic bf16 WMMA GEMM: C[M,N] = A[Mpad,K](bf16) * Bw[N,K](bf16, pre-T) + bias
// block: 256 thr = 8 waves (2x4), tile 128x128, K-step 32. Wave tile 64x32.
// Double-buffered LDS, filled with async global->LDS loads (prefetch depth 1).
// MODE 0: qkv scatter (q/k row-major, v transposed)   MODE 1/3: fp32 out
// MODE 2: relu -> bf16 out
// ---------------------------------------------------------------------------
template <int MODE>
__global__ __launch_bounds__(256) void gemm_bf16_k(
    const bf16_t* __restrict__ A, const bf16_t* __restrict__ Bw,
    const float* __restrict__ bias, int K, int N,
    float* __restrict__ fout, bf16_t* __restrict__ bfout,
    bf16_t* __restrict__ qout, bf16_t* __restrict__ kout,
    bf16_t* __restrict__ vout) {
  constexpr int LDA = 40;  // 32 + 8 pad, keeps 16B alignment
  constexpr int LDB = 40;
  __shared__ __align__(16) bf16_t As[2][128 * LDA];
  __shared__ __align__(16) bf16_t Bs[2][128 * LDB];

  const int tid  = threadIdx.x;
  const int wave = tid >> 5, lane = tid & 31;
  const int wm = wave >> 2, wn = wave & 3;  // 2 x 4 wave grid
  const int lh = lane & 15;
  const bool hiL = lane >= 16;
  const int mbase = blockIdx.x * 128, nbase = blockIdx.y * 128;
  const int srow = tid >> 1, scol = (tid & 1) * 16;

  // per-tile async stage: 4 async b128 per thread (2 for A, 2 for B)
  auto stage = [&](int buf, int k0) {
    const bf16_t* ga = A + (size_t)(mbase + srow) * K + k0 + scol;
    async_ld_b128(&As[buf][srow * LDA + scol], ga);
    async_ld_b128(&As[buf][srow * LDA + scol + 8], ga + 8);
    const bf16_t* gb = Bw + (size_t)(nbase + srow) * K + k0 + scol;
    async_ld_b128(&Bs[buf][srow * LDB + scol], gb);
    async_ld_b128(&Bs[buf][srow * LDB + scol + 8], gb + 8);
  };

  v8f acc[4][2];
#pragma unroll
  for (int i = 0; i < 4; i++)
#pragma unroll
    for (int j = 0; j < 2; j++) acc[i][j] = (v8f){0, 0, 0, 0, 0, 0, 0, 0};

  const int nk = K >> 5;
  stage(0, 0);
  for (int kt = 0; kt < nk; ++kt) {
    const int cur = kt & 1;
    if (kt + 1 < nk) {
      stage(cur ^ 1, (kt + 1) << 5);
      ASYNC_WAIT(4);  // previous tile landed; next tile still in flight
    } else {
      ASYNC_WAIT(0);
    }
    __syncthreads();

    const bf16_t* as = As[cur];
    const bf16_t* bs = Bs[cur];
    // A frag (16x32): lanes<16 rows K{0..7,16..23}, lanes>=16 K{8..15,24..31}
    v16bf af[4], bfr[2];
    const int ka = hiL ? 8 : 0;
#pragma unroll
    for (int mi = 0; mi < 4; mi++) {
      const bf16_t* p = as + (wm * 64 + mi * 16 + lh) * LDA + ka;
      af[mi] = ldpair(p, p + 16);
    }
    // B frag (32x16): lanes<16 K{0..7,16..23}, lanes>=16 K{8..15,24..31}
    const int kb = hiL ? 8 : 0;
#pragma unroll
    for (int ni = 0; ni < 2; ni++) {
      const bf16_t* p = bs + (wn * 32 + ni * 16 + lh) * LDB + kb;
      bfr[ni] = ldpair(p, p + 16);
    }
#pragma unroll
    for (int mi = 0; mi < 4; mi++)
#pragma unroll
      for (int ni = 0; ni < 2; ni++)
        acc[mi][ni] = wmma_bf16(af[mi], bfr[ni], acc[mi][ni]);

    __syncthreads();  // tile fully consumed; safe to overwrite next round
  }

  // epilogue: C vgpr r <-> row r (lanes<16) / r+8 (lanes>=16), col = lane%16
#pragma unroll
  for (int mi = 0; mi < 4; mi++) {
#pragma unroll
    for (int ni = 0; ni < 2; ni++) {
      const int colg = nbase + wn * 32 + ni * 16 + lh;
      const float bv = bias[colg];
#pragma unroll
      for (int r = 0; r < 8; r++) {
        const int rowg = mbase + wm * 64 + mi * 16 + r + (hiL ? 8 : 0);
        const float v = acc[mi][ni][r] + bv;
        if (MODE == 0) {
          if (rowg < MROWS) {
            const int b  = rowg / S_LEN;
            const int s  = rowg - b * S_LEN;
            const int wh = colg >> 9;           // 0:q 1:k 2:v
            const int h  = (colg >> 8) & 1;
            const int hd = colg & 255;
            const int bh = b * NHEAD + h;
            if (wh == 0)
              qout[((size_t)bh * SQP + s) * HDIM + hd] = (bf16_t)v;
            else if (wh == 1)
              kout[((size_t)bh * SQP + s) * HDIM + hd] = (bf16_t)v;
            else
              vout[((size_t)bh * HDIM + hd) * SQP + s] = (bf16_t)v;  // V^T
          }
        } else if (MODE == 1 || MODE == 3) {
          fout[(size_t)rowg * N + colg] = v;
        } else {  // MODE 2
          bfout[(size_t)rowg * N + colg] = (bf16_t)fmaxf(v, 0.f);
        }
      }
    }
  }
}

// ---------------------------------------------------------------------------
// causal flash attention, bf16 WMMA, HD=256, block = (b,h, 128 q rows)
// wave owns 16 q rows: Q frags + O accumulator in registers, K/V staged in
// LDS via async global->LDS loads.
// ---------------------------------------------------------------------------
__global__ __launch_bounds__(256) void attn_k(const bf16_t* __restrict__ Q,
                                              const bf16_t* __restrict__ Km,
                                              const bf16_t* __restrict__ Vt,
                                              bf16_t* __restrict__ Obf) {
  constexpr int LDK = HDIM + 8;  // 264
  constexpr int LDV = 40;
  constexpr int LDP = 40;
  __shared__ __align__(16) bf16_t Ks[32 * LDK];    // 32 keys x 256 hd
  __shared__ __align__(16) bf16_t Vs[HDIM * LDV];  // 256 hd x 32 kv (V^T rows)
  __shared__ __align__(16) bf16_t Ps[8 * 16 * LDP];

  const int qb = blockIdx.x, bh = blockIdx.y;
  const int tid  = threadIdx.x;
  const int wave = tid >> 5, lane = tid & 31;
  const int lh = lane & 15;
  const bool hiL = lane >= 16;
  const bf16_t* qp = Q + (size_t)bh * SQP * HDIM;
  const bf16_t* kp = Km + (size_t)bh * SQP * HDIM;
  const bf16_t* vp = Vt + (size_t)bh * HDIM * SQP;
  const int qrow0 = qb * 128 + wave * 16;

  // Q fragments, persistent (8 k-steps of 32 over HD=256)
  v16bf qf[8];
  {
    const bf16_t* qr = qp + (size_t)(qrow0 + lh) * HDIM + (hiL ? 8 : 0);
#pragma unroll
    for (int ks = 0; ks < 8; ks++) qf[ks] = ldpair(qr + ks * 32, qr + ks * 32 + 16);
  }

  v8f o[16];
#pragma unroll
  for (int j = 0; j < 16; j++) o[j] = (v8f){0, 0, 0, 0, 0, 0, 0, 0};
  float m[8], l[8];
#pragma unroll
  for (int r = 0; r < 8; r++) { m[r] = -__builtin_inff(); l[r] = 0.f; }

  const int kvend = min(S_LEN, qb * 128 + 128);
  const int nt = (kvend + 31) >> 5;

  for (int t = 0; t < nt; ++t) {
    const int kv0 = t * 32;
    __syncthreads();  // previous tile fully consumed by all waves
    {  // async stage K block (32 x 256) and V^T block (256 x 32)
      const int key = tid >> 3, ch = (tid & 7) * 32;
      const bf16_t* g = kp + (size_t)(kv0 + key) * HDIM + ch;
      bf16_t* d = Ks + key * LDK + ch;
      async_ld_b128(d, g);
      async_ld_b128(d + 8, g + 8);
      async_ld_b128(d + 16, g + 16);
      async_ld_b128(d + 24, g + 24);
      const bf16_t* gv = vp + (size_t)tid * SQP + kv0;
      bf16_t* dv = Vs + tid * LDV;
      async_ld_b128(dv, gv);
      async_ld_b128(dv + 8, gv + 8);
      async_ld_b128(dv + 16, gv + 16);
      async_ld_b128(dv + 24, gv + 24);
    }
    ASYNC_WAIT(0);
    __syncthreads();

    // scores: two 16x16 tiles (32 keys), accumulate over HD
    v8f c0 = (v8f){0, 0, 0, 0, 0, 0, 0, 0};
    v8f c1 = (v8f){0, 0, 0, 0, 0, 0, 0, 0};
    const int kb = hiL ? 8 : 0;
#pragma unroll
    for (int ks = 0; ks < 8; ks++) {
      const bf16_t* p0 = Ks + lh * LDK + ks * 32 + kb;
      const bf16_t* p1 = Ks + (16 + lh) * LDK + ks * 32 + kb;
      v16bf b0 = ldpair(p0, p0 + 16);
      v16bf b1 = ldpair(p1, p1 + 16);
      c0 = wmma_bf16(qf[ks], b0, c0);
      c1 = wmma_bf16(qf[ks], b1, c1);
    }

    // online softmax in C-tile layout (stats replicated per 16-lane half)
    float alpha[8];
#pragma unroll
    for (int r = 0; r < 8; r++) {
      const int rowg = qrow0 + r + (hiL ? 8 : 0);
      const int col0 = kv0 + lh, col1 = kv0 + 16 + lh;
      const float s0 =
          (col0 <= rowg && col0 < S_LEN) ? c0[r] * 0.0625f : -__builtin_inff();
      const float s1 =
          (col1 <= rowg && col1 < S_LEN) ? c1[r] * 0.0625f : -__builtin_inff();
      const float mx = half16_max(fmaxf(s0, s1));
      const float mn = fmaxf(m[r], mx);
      const float al = __expf(m[r] - mn);
      const float e0 = __expf(s0 - mn);
      const float e1 = __expf(s1 - mn);
      const float rs = half16_sum(e0 + e1);
      m[r] = mn;
      l[r] = l[r] * al + rs;
      alpha[r] = al;
      c0[r] = e0;
      c1[r] = e1;
    }
#pragma unroll
    for (int j = 0; j < 16; j++)
#pragma unroll
      for (int r = 0; r < 8; r++) o[j][r] *= alpha[r];

    // re-layout exp(P): C layout -> A layout through per-wave LDS tile
    bf16_t* pw = Ps + wave * 16 * LDP;
#pragma unroll
    for (int r = 0; r < 8; r++) {
      const int pr = r + (hiL ? 8 : 0);
      pw[pr * LDP + lh]      = (bf16_t)c0[r];
      pw[pr * LDP + 16 + lh] = (bf16_t)c1[r];
    }
    const bf16_t* pa0 = pw + lh * LDP + (hiL ? 8 : 0);
    v16bf pa = ldpair(pa0, pa0 + 16);

    // O(16x256) += P(16x32) @ V(32x256); V^T rows give contiguous B frags
#pragma unroll
    for (int j = 0; j < 16; j++) {
      const bf16_t* vb = Vs + (j * 16 + lh) * LDV + kb;
      v16bf bv = ldpair(vb, vb + 16);
      o[j] = wmma_bf16(pa, bv, o[j]);
    }
  }

  // normalize and store bf16 into [Mpad, D] activation for the Wo GEMM
  const int b = bh >> 1, h = bh & 1;
#pragma unroll
  for (int r = 0; r < 8; r++) l[r] = 1.f / l[r];
#pragma unroll
  for (int r = 0; r < 8; r++) {
    const int srow = qrow0 + r + (hiL ? 8 : 0);
    if (srow < S_LEN) {
      bf16_t* op = Obf + ((size_t)(b * S_LEN + srow)) * DMODEL + h * HDIM + lh;
#pragma unroll
      for (int j = 0; j < 16; j++) op[j * 16] = (bf16_t)(o[j][r] * l[r]);
    }
  }
}

// ---------------------------------------------------------------------------
// fused residual + LayerNorm (one row per block)
// ---------------------------------------------------------------------------
__global__ __launch_bounds__(256) void ln_k(const float* __restrict__ pre,
                                            const float* __restrict__ res,
                                            const float* __restrict__ g,
                                            const float* __restrict__ bta,
                                            float* __restrict__ fo,
                                            bf16_t* __restrict__ bfo) {
  const int row = blockIdx.x, tid = threadIdx.x;
  const size_t base = (size_t)row * DMODEL;
  const float x0 = pre[base + tid] + res[base + tid];
  const float x1 = pre[base + tid + 256] + res[base + tid + 256];
  float s = x0 + x1, s2 = x0 * x0 + x1 * x1;
#pragma unroll
  for (int msk = 1; msk < 32; msk <<= 1) {
    s += __shfl_xor(s, msk, 32);
    s2 += __shfl_xor(s2, msk, 32);
  }
  __shared__ float sh[16];
  const int wave = tid >> 5, lane = tid & 31;
  if (lane == 0) { sh[wave] = s; sh[8 + wave] = s2; }
  __syncthreads();
  float ts = 0.f, ts2 = 0.f;
#pragma unroll
  for (int i = 0; i < 8; i++) { ts += sh[i]; ts2 += sh[8 + i]; }
  const float mean = ts * (1.f / DMODEL);
  const float var = ts2 * (1.f / DMODEL) - mean * mean;
  const float rstd = rsqrtf(var + 1e-5f);
  const float y0 = (x0 - mean) * rstd * g[tid] + bta[tid];
  const float y1 = (x1 - mean) * rstd * g[tid + 256] + bta[tid + 256];
  fo[base + tid] = y0;
  fo[base + tid + 256] = y1;
  if (bfo) {
    bfo[base + tid] = (bf16_t)y0;
    bfo[base + tid + 256] = (bf16_t)y1;
  }
}

// ---------------------------------------------------------------------------
// conversion kernels
// ---------------------------------------------------------------------------
__global__ __launch_bounds__(256) void cvt_x_k(const float* __restrict__ x,
                                               bf16_t* __restrict__ a0) {
  const size_t i = (size_t)blockIdx.x * 256 + threadIdx.x;
  const size_t valid = (size_t)MROWS * DMODEL;
  a0[i] = (i < valid) ? (bf16_t)x[i] : (bf16_t)0.f;
}

__global__ __launch_bounds__(256) void cvt_wT_k(const float* __restrict__ w,
                                                bf16_t* __restrict__ wt, int K,
                                                int N) {
  const size_t i = (size_t)blockIdx.x * 256 + threadIdx.x;
  if (i >= (size_t)N * K) return;
  const int n = (int)(i / (size_t)K);
  const int k = (int)(i - (size_t)n * K);
  wt[i] = (bf16_t)w[(size_t)k * N + n];  // [K,N] -> [N,K]
}

// ---------------------------------------------------------------------------
// launcher
// ---------------------------------------------------------------------------
extern "C" void kernel_launch(void* const* d_in, const int* in_sizes, int n_in,
                              void* d_out, int out_size, void* d_ws,
                              size_t ws_size, hipStream_t stream) {
  (void)in_sizes; (void)n_in; (void)out_size; (void)ws_size;
  const float* x    = (const float*)d_in[0];
  const float* Wqkv = (const float*)d_in[1];
  const float* bqkv = (const float*)d_in[2];
  const float* Wo   = (const float*)d_in[3];
  const float* bo   = (const float*)d_in[4];
  const float* g1   = (const float*)d_in[5];
  const float* be1  = (const float*)d_in[6];
  const float* W1   = (const float*)d_in[7];
  const float* b1f  = (const float*)d_in[8];
  const float* W2   = (const float*)d_in[9];
  const float* b2f  = (const float*)d_in[10];
  const float* g2   = (const float*)d_in[11];
  const float* be2  = (const float*)d_in[12];

  char* ws = (char*)d_ws;
  size_t off = 0;
  auto alloc = [&](size_t bytes) -> char* {
    char* p = ws + off;
    off = (off + bytes + 255) & ~(size_t)255;
    return p;
  };
  bf16_t* A0    = (bf16_t*)alloc((size_t)MPAD * DMODEL * 2);
  bf16_t* WqkvT = (bf16_t*)alloc((size_t)3 * DMODEL * DMODEL * 2);
  bf16_t* WoT   = (bf16_t*)alloc((size_t)DMODEL * DMODEL * 2);
  bf16_t* W1T   = (bf16_t*)alloc((size_t)FFDIM * DMODEL * 2);
  bf16_t* W2T   = (bf16_t*)alloc((size_t)DMODEL * FFDIM * 2);
  bf16_t* Qb    = (bf16_t*)alloc((size_t)BHN * SQP * HDIM * 2);
  bf16_t* Kb    = (bf16_t*)alloc((size_t)BHN * SQP * HDIM * 2);
  bf16_t* Vtb   = (bf16_t*)alloc((size_t)BHN * HDIM * SQP * 2);
  bf16_t* Ob    = (bf16_t*)alloc((size_t)MPAD * DMODEL * 2);
  bf16_t* Hb    = (bf16_t*)alloc((size_t)MPAD * DMODEL * 2);
  bf16_t* F1b   = (bf16_t*)alloc((size_t)MPAD * FFDIM * 2);
  float*  Tmp   = (float*)alloc((size_t)MPAD * DMODEL * 4);  // reused for FFN2
  float*  Hf    = (float*)alloc((size_t)MPAD * DMODEL * 4);

  // V^T must be zero in the padded key range (0 * V_pad must stay 0)
  hipMemsetAsync(Vtb, 0, (size_t)BHN * HDIM * SQP * 2, stream);

  cvt_x_k<<<(MPAD * DMODEL) / 256, 256, 0, stream>>>(x, A0);
  cvt_wT_k<<<(3 * DMODEL * DMODEL) / 256, 256, 0, stream>>>(Wqkv, WqkvT, DMODEL,
                                                            3 * DMODEL);
  cvt_wT_k<<<(DMODEL * DMODEL) / 256, 256, 0, stream>>>(Wo, WoT, DMODEL, DMODEL);
  cvt_wT_k<<<(FFDIM * DMODEL) / 256, 256, 0, stream>>>(W1, W1T, DMODEL, FFDIM);
  cvt_wT_k<<<(DMODEL * FFDIM) / 256, 256, 0, stream>>>(W2, W2T, FFDIM, DMODEL);

  // 1) QKV projection + head split
  gemm_bf16_k<0><<<dim3(MPAD / 128, (3 * DMODEL) / 128), 256, 0, stream>>>(
      A0, WqkvT, bqkv, DMODEL, 3 * DMODEL, nullptr, nullptr, Qb, Kb, Vtb);
  // 2) causal flash attention
  attn_k<<<dim3(SQP / 128, BHN), 256, 0, stream>>>(Qb, Kb, Vtb, Ob);
  // 3) output projection
  gemm_bf16_k<1><<<dim3(MPAD / 128, DMODEL / 128), 256, 0, stream>>>(
      Ob, WoT, bo, DMODEL, DMODEL, Tmp, nullptr, nullptr, nullptr, nullptr);
  // 4) residual + LN1 -> h (fp32 + bf16)
  ln_k<<<MROWS, 256, 0, stream>>>(Tmp, x, g1, be1, Hf, Hb);
  // 5) FFN up + ReLU
  gemm_bf16_k<2><<<dim3(MPAD / 128, FFDIM / 128), 256, 0, stream>>>(
      Hb, W1T, b1f, DMODEL, FFDIM, nullptr, F1b, nullptr, nullptr, nullptr);
  // 6) FFN down
  gemm_bf16_k<3><<<dim3(MPAD / 128, DMODEL / 128), 256, 0, stream>>>(
      F1b, W2T, b2f, FFDIM, DMODEL, Tmp, nullptr, nullptr, nullptr, nullptr);
  // 7) residual + LN2 -> output
  ln_k<<<MROWS, 256, 0, stream>>>(Tmp, Hf, g2, be2, (float*)d_out, nullptr);
}
